// RopeBlock_67886253080710
// MI455X (gfx1250) — matrix operand order repaired and checked by reading, not compiled, not run on hardware
//
#include <hip/hip_runtime.h>
#include <hip/hip_bf16.h>

// ---------------------------------------------------------------------------
// Transformer block for MI455X (gfx1250, wave32, WMMA).
// fp32 I/O; all GEMM contractions run on v_wmma_f32_16x16x32_bf16.
// ---------------------------------------------------------------------------

typedef __attribute__((ext_vector_type(16))) __bf16 v16bf;
typedef __attribute__((ext_vector_type(8)))  float  v8f;
typedef __attribute__((ext_vector_type(4)))  int    v4i_;

static constexpr int B_   = 2;
static constexpr int N_   = 2048;
static constexpr int C_   = 1024;
static constexpr int H_   = 16;
static constexpr int D_   = 64;
static constexpr int HID_ = 4096;
static constexpr int M_   = B_ * N_;   // 4096 rows

// gfx1250 async global->LDS copy path (guarded: only if this clang has it)
#if __has_builtin(__builtin_amdgcn_global_load_async_to_lds_b128) && \
    __has_builtin(__builtin_amdgcn_s_wait_asynccnt)
#define USE_ASYNC_COPY 1
#else
#define USE_ASYNC_COPY 0
#endif

#if USE_ASYNC_COPY
// builtin signature: (v4i addrspace(1)*, v4i addrspace(3)*, imm offset, imm cpol)
#define AS1P(p) ((__attribute__((address_space(1))) v4i_*)(p))
#define AS3P(p) ((__attribute__((address_space(3))) v4i_*)(p))
#endif

// ---------------------------------------------------------------------------
// fp32 -> bf16 conversion (4 elems / thread)
// ---------------------------------------------------------------------------
__global__ __launch_bounds__(256) void cvt_bf16_kernel(const float* __restrict__ in,
                                                       __bf16* __restrict__ out, int n) {
  int i = (blockIdx.x * 256 + threadIdx.x) * 4;
  if (i < n) {
    float4 v = *(const float4*)(in + i);
    out[i + 0] = (__bf16)v.x;
    out[i + 1] = (__bf16)v.y;
    out[i + 2] = (__bf16)v.z;
    out[i + 3] = (__bf16)v.w;
  }
}

// ---------------------------------------------------------------------------
// LayerNorm over C=1024, one row per block, bf16 out
// ---------------------------------------------------------------------------
__global__ __launch_bounds__(256) void layernorm_bf16_kernel(const float* __restrict__ x,
                                                             const float* __restrict__ g,
                                                             const float* __restrict__ bta,
                                                             __bf16* __restrict__ out) {
  __shared__ float s1[256], s2[256];
  const int row = blockIdx.x;
  const float* xr = x + (size_t)row * C_;
  float vals[4], a0 = 0.f, a1 = 0.f;
#pragma unroll
  for (int i = 0; i < 4; ++i) {
    float v = xr[threadIdx.x + i * 256];
    vals[i] = v; a0 += v; a1 += v * v;
  }
  s1[threadIdx.x] = a0; s2[threadIdx.x] = a1;
  __syncthreads();
  for (int st = 128; st > 0; st >>= 1) {
    if (threadIdx.x < st) {
      s1[threadIdx.x] += s1[threadIdx.x + st];
      s2[threadIdx.x] += s2[threadIdx.x + st];
    }
    __syncthreads();
  }
  const float mu   = s1[0] * (1.f / C_);
  const float var  = s2[0] * (1.f / C_) - mu * mu;
  const float rstd = rsqrtf(var + 1e-5f);
#pragma unroll
  for (int i = 0; i < 4; ++i) {
    int c = threadIdx.x + i * 256;
    out[(size_t)row * C_ + c] = (__bf16)((vals[i] - mu) * rstd * g[c] + bta[c]);
  }
}

// ---------------------------------------------------------------------------
// Tiled WMMA GEMM: D[M x Nn] = A[M x K](bf16) * W[K x Nn](bf16) + bias
// block tile 128x128, 8 waves in 2x4, each wave 64x32 (4x2 wmma accums).
// Double-buffered LDS pipeline; async global->LDS copies if available.
// Epilogue: optional GELU / residual; fp32 or bf16 out.
// ---------------------------------------------------------------------------
template <int OUT_F32, int GELU, int RESID>
__global__ __launch_bounds__(256)
void gemm_bf16_kernel(const __bf16* __restrict__ A, const __bf16* __restrict__ Wt,
                      const float* __restrict__ bias, const float* __restrict__ resid,
                      float* __restrict__ outF, __bf16* __restrict__ outB,
                      int M, int Nn, int K) {
  constexpr int ASTR = 40;   // 32 + 8 pad (bf16), keeps 16B alignment
  constexpr int BSTR = 136;  // 128 + 8 pad
  __shared__ __align__(16) __bf16 As[2][128 * ASTR];
  __shared__ __align__(16) __bf16 Bs[2][32 * BSTR];

  const int t    = threadIdx.x;
  const int wave = t >> 5, lane = t & 31;
  const int half = lane >> 4, ln = lane & 15;
  const int wm = wave >> 2, wn = wave & 3;       // 2 x 4 wave grid
  const int m0 = blockIdx.y * 128, n0 = blockIdx.x * 128;

  // per-thread cooperative-copy coordinates (16B chunks)
  const int arow = t >> 2,  acol = (t & 3) * 8;   // A: rows arow, arow+64
  const int brow = t >> 4,  bcol = (t & 15) * 8;  // B: rows brow, brow+16

  v8f acc[4][2] = {};

  auto compute = [&](int buf) {
    // B fragments: (k,n) -> k = (e/8)*16 + half*8 + e%8, n = ln
    v16bf bfrag[2];
#pragma unroll
    for (int nt = 0; nt < 2; ++nt)
#pragma unroll
      for (int e = 0; e < 16; ++e)
        bfrag[nt][e] = Bs[buf][((e >> 3) * 16 + half * 8 + (e & 7)) * BSTR + wn * 32 + nt * 16 + ln];
#pragma unroll
    for (int mt = 0; mt < 4; ++mt) {
      // A fragment: m = ln, k = (e/8)*16 + half*8 + e%8 -> two contiguous 16B reads
      v16bf afrag;
      const int mrow = (wm * 64 + mt * 16 + ln) * ASTR;
      *(uint4*)&afrag       = *(const uint4*)&As[buf][mrow + half * 8];
      *((uint4*)&afrag + 1) = *(const uint4*)&As[buf][mrow + 16 + half * 8];
#pragma unroll
      for (int nt = 0; nt < 2; ++nt)
        acc[mt][nt] = __builtin_amdgcn_wmma_f32_16x16x32_bf16(
            false, afrag, false, bfrag[nt], (short)0, acc[mt][nt], false, false);
    }
  };

  const int nk = K / 32;

#if USE_ASYNC_COPY
  auto stage = [&](int buf, int k0) {
    __builtin_amdgcn_global_load_async_to_lds_b128(
        AS1P(&A[(size_t)(m0 + arow) * K + k0 + acol]),
        AS3P(&As[buf][arow * ASTR + acol]), 0, 0);
    __builtin_amdgcn_global_load_async_to_lds_b128(
        AS1P(&A[(size_t)(m0 + arow + 64) * K + k0 + acol]),
        AS3P(&As[buf][(arow + 64) * ASTR + acol]), 0, 0);
    __builtin_amdgcn_global_load_async_to_lds_b128(
        AS1P(&Wt[(size_t)(k0 + brow) * Nn + n0 + bcol]),
        AS3P(&Bs[buf][brow * BSTR + bcol]), 0, 0);
    __builtin_amdgcn_global_load_async_to_lds_b128(
        AS1P(&Wt[(size_t)(k0 + brow + 16) * Nn + n0 + bcol]),
        AS3P(&Bs[buf][(brow + 16) * BSTR + bcol]), 0, 0);
  };
  stage(0, 0);
  __builtin_amdgcn_s_wait_asynccnt(0);
  __syncthreads();
  for (int kk = 0; kk < nk; ++kk) {
    if (kk + 1 < nk) stage((kk + 1) & 1, (kk + 1) * 32);
    compute(kk & 1);
    if (kk + 1 < nk) __builtin_amdgcn_s_wait_asynccnt(0);
    __syncthreads();
  }
#else
  uint4 ra0, ra1, rb0, rb1;
  auto gload = [&](int k0) {
    ra0 = *(const uint4*)&A[(size_t)(m0 + arow) * K + k0 + acol];
    ra1 = *(const uint4*)&A[(size_t)(m0 + arow + 64) * K + k0 + acol];
    rb0 = *(const uint4*)&Wt[(size_t)(k0 + brow) * Nn + n0 + bcol];
    rb1 = *(const uint4*)&Wt[(size_t)(k0 + brow + 16) * Nn + n0 + bcol];
  };
  auto sstore = [&](int buf) {
    *(uint4*)&As[buf][arow * ASTR + acol]        = ra0;
    *(uint4*)&As[buf][(arow + 64) * ASTR + acol] = ra1;
    *(uint4*)&Bs[buf][brow * BSTR + bcol]        = rb0;
    *(uint4*)&Bs[buf][(brow + 16) * BSTR + bcol] = rb1;
  };
  gload(0);
  sstore(0);
  __syncthreads();
  for (int kk = 0; kk < nk; ++kk) {
    if (kk + 1 < nk) gload((kk + 1) * 32);   // prefetch into regs, overlaps compute
    compute(kk & 1);
    if (kk + 1 < nk) sstore((kk + 1) & 1);
    __syncthreads();
  }
#endif

  // epilogue
#pragma unroll
  for (int mt = 0; mt < 4; ++mt)
#pragma unroll
    for (int nt = 0; nt < 2; ++nt)
#pragma unroll
      for (int r = 0; r < 8; ++r) {
        const int gm = m0 + wm * 64 + mt * 16 + r + 8 * half;
        const int gn = n0 + wn * 32 + nt * 16 + ln;
        float v = acc[mt][nt][r] + bias[gn];
        if (GELU)  v = 0.5f * v * (1.0f + erff(v * 0.70710678118654752f));
        if (RESID) v += resid[(size_t)gm * Nn + gn];
        if (OUT_F32) outF[(size_t)gm * Nn + gn] = v;
        else         outB[(size_t)gm * Nn + gn] = (__bf16)v;
      }
}

// ---------------------------------------------------------------------------
// RoPE + split: qkv fp32 (B,N,3C) -> q,k (roped), v as bf16 (B,H,N,D)
// ---------------------------------------------------------------------------
__global__ __launch_bounds__(256) void rope_split_kernel(const float* __restrict__ qkv,
                                                         __bf16* __restrict__ qb,
                                                         __bf16* __restrict__ kb,
                                                         __bf16* __restrict__ vb) {
  const int bn = blockIdx.x;           // b*N + n
  const int b = bn / N_, n = bn % N_;
  const float* row = qkv + (size_t)bn * 3 * C_;
  const float pos = (float)n;
  // q,k pairs: 2 * H * (D/2) = 1024
  for (int idx = threadIdx.x; idx < 1024; idx += 256) {
    const int s = idx >> 9;            // 0=q, 1=k
    const int rem = idx & 511;
    const int h = rem >> 5, i = rem & 31;
    const float x1 = row[s * C_ + h * D_ + 2 * i];
    const float x2 = row[s * C_ + h * D_ + 2 * i + 1];
    const float fr = pos * __expf(-0.28782313662f * (float)i);  // ln(1e4)/32
    const float sn = __sinf(fr), cs = __cosf(fr);
    __bf16* dst = (s == 0 ? qb : kb);
    const size_t o = ((size_t)(b * H_ + h) * N_ + n) * D_ + 2 * i;
    dst[o]     = (__bf16)(x1 * cs - x2 * sn);
    dst[o + 1] = (__bf16)(x1 * sn + x2 * cs);
  }
  // v: H*D = 1024 plain copies
  for (int idx = threadIdx.x; idx < 1024; idx += 256) {
    const int h = idx >> 6, d = idx & 63;
    vb[((size_t)(b * H_ + h) * N_ + n) * D_ + d] = (__bf16)row[2 * C_ + h * D_ + d];
  }
}

// ---------------------------------------------------------------------------
// Flash attention. Per block: 128 query rows (16 per wave), all waves share
// a 32-key K/V tile staged in LDS (V stored transposed so B-fragments read
// contiguously). S via 2 wmma k-steps over D=64; online softmax fp32;
// P through LDS to A-frag layout; O accumulated with 4 wmma per key step.
// ---------------------------------------------------------------------------
__global__ __launch_bounds__(256) void attention_kernel(const __bf16* __restrict__ Q,
                                                        const __bf16* __restrict__ Km,
                                                        const __bf16* __restrict__ V,
                                                        __bf16* __restrict__ O) {
  constexpr int KSTR = 72;  // 64 + 8 pad
  constexpr int VSTR = 40;  // 32 + 8 pad
  __shared__ __align__(16) __bf16 Ks[32 * KSTR];   // [key][dim]
  __shared__ __align__(16) __bf16 Vs[64 * VSTR];   // [dim][key]  (transposed)
  __shared__ __align__(16) __bf16 pbuf[8][16][32];

  const int bh = blockIdx.x;
  const int b = bh / H_, h = bh % H_;
  const int wave = threadIdx.x >> 5, lane = threadIdx.x & 31;
  const int half = lane >> 4, ln = lane & 15;
  const int q0 = blockIdx.y * 128 + wave * 16;
  const size_t base = (size_t)bh * N_ * D_;
  const float scale = 0.125f;  // D^-0.5

  // staging coords: one 16B chunk per thread per tile
  const int skd = threadIdx.x >> 3;          // key 0..31
  const int sdb = (threadIdx.x & 7) * 8;     // dim base 0..56

  // Q fragments (16 x 64 -> two 16x32 A-frags), held in registers throughout
  v16bf qf[2];
#pragma unroll
  for (int t = 0; t < 2; ++t) {
    const __bf16* qrow = Q + base + (size_t)(q0 + ln) * D_ + t * 32;
    *(uint4*)&qf[t]       = *(const uint4*)(qrow + half * 8);
    *((uint4*)&qf[t] + 1) = *(const uint4*)(qrow + 16 + half * 8);
  }

  v8f accO[4] = {};
  float mrun[8], lrun[8];
#pragma unroll
  for (int r = 0; r < 8; ++r) { mrun[r] = -1e30f; lrun[r] = 0.f; }

  for (int kb = 0; kb < N_; kb += 32) {
    __syncthreads();  // previous iteration's tile reads complete
    {
      const size_t grow = base + (size_t)(kb + skd) * D_ + sdb;
      *(uint4*)&Ks[skd * KSTR + sdb] = *(const uint4*)(Km + grow);
      union { uint4 u; __bf16 hh[8]; } vt;
      vt.u = *(const uint4*)(V + grow);
#pragma unroll
      for (int j = 0; j < 8; ++j) Vs[(sdb + j) * VSTR + skd] = vt.hh[j];
    }
    __syncthreads();  // tiles ready

    // S = Q K^T for 32 keys -> two 16x16 fp32 tiles
    v8f s[2] = {};
#pragma unroll
    for (int t = 0; t < 2; ++t)
#pragma unroll
      for (int j = 0; j < 2; ++j) {
        const __bf16* kr = &Ks[(j * 16 + ln) * KSTR + t * 32 + half * 8];
        v16bf kf;
        *(uint4*)&kf       = *(const uint4*)kr;
        *((uint4*)&kf + 1) = *(const uint4*)(kr + 16);
        s[j] = __builtin_amdgcn_wmma_f32_16x16x32_bf16(
            false, qf[t], false, kf, (short)0, s[j], false, false);
      }

    // online softmax (rows live in 16-lane halves of the C layout)
#pragma unroll
    for (int r = 0; r < 8; ++r) {
      float v0 = s[0][r] * scale, v1 = s[1][r] * scale;
      float m = fmaxf(v0, v1);
#pragma unroll
      for (int off = 8; off >= 1; off >>= 1)
        m = fmaxf(m, __shfl_xor(m, off, 16));
      const float nm = fmaxf(mrun[r], m);
      const float alpha = __expf(mrun[r] - nm);
      mrun[r] = nm;
      const float p0 = __expf(v0 - nm), p1 = __expf(v1 - nm);
      s[0][r] = p0; s[1][r] = p1;
      float rs = p0 + p1;
#pragma unroll
      for (int off = 8; off >= 1; off >>= 1)
        rs += __shfl_xor(rs, off, 16);
      lrun[r] = lrun[r] * alpha + rs;
#pragma unroll
      for (int d = 0; d < 4; ++d) accO[d][r] *= alpha;
    }

    // P (C layout) -> LDS -> A-frag layout
#pragma unroll
    for (int r = 0; r < 8; ++r) {
      pbuf[wave][r + 8 * half][ln]      = (__bf16)s[0][r];
      pbuf[wave][r + 8 * half][16 + ln] = (__bf16)s[1][r];
    }
    __syncthreads();
    v16bf pf;
    {
      const __bf16* prow = &pbuf[wave][ln][0];
      *(uint4*)&pf       = *(const uint4*)(prow + half * 8);
      *((uint4*)&pf + 1) = *(const uint4*)(prow + 16 + half * 8);
    }

    // O += P @ V : V B-frag (k = key-in-32 contiguous in Vs rows, n = dim col)
#pragma unroll
    for (int d = 0; d < 4; ++d) {
      const __bf16* vr = &Vs[(d * 16 + ln) * VSTR + half * 8];
      v16bf vf;
      *(uint4*)&vf       = *(const uint4*)vr;
      *((uint4*)&vf + 1) = *(const uint4*)(vr + 16);
      accO[d] = __builtin_amdgcn_wmma_f32_16x16x32_bf16(
          false, pf, false, vf, (short)0, accO[d], false, false);
    }
  }

  // normalize + write (B,N,C) bf16
#pragma unroll
  for (int d = 0; d < 4; ++d)
#pragma unroll
    for (int r = 0; r < 8; ++r) {
      const int qm = q0 + r + 8 * half;
      const float v = accO[d][r] / lrun[r];
      O[((size_t)(b * N_ + qm)) * C_ + h * D_ + d * 16 + ln] = (__bf16)v;
    }
}

// ---------------------------------------------------------------------------
// host launcher
// ---------------------------------------------------------------------------
extern "C" void kernel_launch(void* const* d_in, const int* in_sizes, int n_in,
                              void* d_out, int out_size, void* d_ws, size_t ws_size,
                              hipStream_t stream) {
  (void)in_sizes; (void)n_in; (void)out_size; (void)ws_size;
  const float* x      = (const float*)d_in[0];
  const float* ln1_g  = (const float*)d_in[1];
  const float* ln1_b  = (const float*)d_in[2];
  const float* W_qkv  = (const float*)d_in[3];
  const float* b_qkv  = (const float*)d_in[4];
  const float* W_proj = (const float*)d_in[5];
  const float* b_proj = (const float*)d_in[6];
  const float* ln2_g  = (const float*)d_in[7];
  const float* ln2_b  = (const float*)d_in[8];
  const float* W_fc1  = (const float*)d_in[9];
  const float* b_fc1  = (const float*)d_in[10];
  const float* W_fc2  = (const float*)d_in[11];
  const float* b_fc2  = (const float*)d_in[12];
  float* out = (float*)d_out;

  char* w = (char*)d_ws;
  auto alloc = [&](size_t bytes) {
    void* p = (void*)w;
    w += (bytes + 255) & ~(size_t)255;
    return p;
  };
  __bf16* wqkv = (__bf16*)alloc((size_t)3 * C_ * C_ * 2);
  __bf16* wprj = (__bf16*)alloc((size_t)C_ * C_ * 2);
  __bf16* wfc1 = (__bf16*)alloc((size_t)C_ * HID_ * 2);
  __bf16* wfc2 = (__bf16*)alloc((size_t)HID_ * C_ * 2);
  __bf16* hn   = (__bf16*)alloc((size_t)M_ * C_ * 2);
  float*  qkvf = (float*) alloc((size_t)M_ * 3 * C_ * 4);
  __bf16* qb   = (__bf16*)alloc((size_t)M_ * C_ * 2);
  __bf16* kb   = (__bf16*)alloc((size_t)M_ * C_ * 2);
  __bf16* vb   = (__bf16*)alloc((size_t)M_ * C_ * 2);
  __bf16* ob   = (__bf16*)alloc((size_t)M_ * C_ * 2);
  float*  x1   = (float*) alloc((size_t)M_ * C_ * 4);
  __bf16* h2   = (__bf16*)alloc((size_t)M_ * C_ * 2);
  __bf16* gb   = (__bf16*)alloc((size_t)M_ * HID_ * 2);

  auto cvt = [&](const float* src, __bf16* dst, int n) {
    cvt_bf16_kernel<<<(n / 4 + 255) / 256, 256, 0, stream>>>(src, dst, n);
  };
  cvt(W_qkv,  wqkv, 3 * C_ * C_);
  cvt(W_proj, wprj, C_ * C_);
  cvt(W_fc1,  wfc1, C_ * HID_);
  cvt(W_fc2,  wfc2, HID_ * C_);

  // --- attention branch ---
  layernorm_bf16_kernel<<<M_, 256, 0, stream>>>(x, ln1_g, ln1_b, hn);
  gemm_bf16_kernel<1, 0, 0><<<dim3(3 * C_ / 128, M_ / 128), 256, 0, stream>>>(
      hn, wqkv, b_qkv, nullptr, qkvf, nullptr, M_, 3 * C_, C_);
  rope_split_kernel<<<B_ * N_, 256, 0, stream>>>(qkvf, qb, kb, vb);
  attention_kernel<<<dim3(B_ * H_, N_ / 128), 256, 0, stream>>>(qb, kb, vb, ob);
  gemm_bf16_kernel<1, 0, 1><<<dim3(C_ / 128, M_ / 128), 256, 0, stream>>>(
      ob, wprj, b_proj, x, x1, nullptr, M_, C_, C_);

  // --- MLP branch ---
  layernorm_bf16_kernel<<<M_, 256, 0, stream>>>(x1, ln2_g, ln2_b, h2);
  gemm_bf16_kernel<0, 1, 0><<<dim3(HID_ / 128, M_ / 128), 256, 0, stream>>>(
      h2, wfc1, b_fc1, nullptr, nullptr, gb, M_, HID_, C_);
  gemm_bf16_kernel<1, 0, 1><<<dim3(C_ / 128, M_ / 128), 256, 0, stream>>>(
      gb, wfc2, b_fc2, x1, out, nullptr, M_, C_, HID_);
}
